// CausalSelfAttention_84482006712872
// MI455X (gfx1250) — matrix operand order repaired and checked by reading, not compiled, not run on hardware
//
#include <hip/hip_runtime.h>
#include <hip/hip_bf16.h>

// ---------------------------------------------------------------------------
// CDNA5 (gfx1250) causal self-attention, bf16 WMMA + fp32 softmax.
// B=4, S=2048, E=1024, H=16, D=64.
//
// All matmul operands are pre-laid-out so every WMMA fragment is a pair of
// contiguous 16-byte global loads (no LDS staging in the GEMMs, no V
// transpose in attention). Weights/activations fit in the 192MB L2, so
// repeated fragment reads are cache hits, not HBM traffic.
// ---------------------------------------------------------------------------

typedef __bf16 bf16_t;
typedef __attribute__((ext_vector_type(16))) __bf16 v16bf;
typedef __attribute__((ext_vector_type(8)))  float  v8f;

#define BATCH 4
#define SEQ   2048
#define EMB   1024
#define HEADS 16
#define HDIM  64
#define ROWS  (BATCH * SEQ)      // 8192

union FragBF {
    v16bf v;
    uint4 q[2];
};

__device__ inline unsigned short f2us(float f) {
    union { float f; unsigned u; } a; a.f = f;
    unsigned u = a.u;
    unsigned r = u + 0x7FFFu + ((u >> 16) & 1u);   // round-to-nearest-even
    return (unsigned short)(r >> 16);
}

__device__ inline bf16_t f2bf(float f) {
    union { unsigned short s; bf16_t b; } o;
    o.s = f2us(f);
    return o.b;
}

// ---------------------------------------------------------------------------
// fp32 -> bf16 elementwise convert (grid-stride), layout preserved
// ---------------------------------------------------------------------------
__global__ void f32_to_bf16_kernel(const float* __restrict__ in,
                                   bf16_t* __restrict__ out, int n) {
    int i = blockIdx.x * blockDim.x + threadIdx.x;
    int stride = gridDim.x * blockDim.x;
    for (; i < n; i += stride) out[i] = f2bf(in[i]);
}

// ---------------------------------------------------------------------------
// fp32 [K][N] -> bf16 [N][K] tiled transpose-convert (coalesced both sides)
// Grid: (K/32, N/32), block 256 (32x8).
// ---------------------------------------------------------------------------
__global__ __launch_bounds__(256)
void transpose_f32_to_bf16_kernel(const float* __restrict__ in,
                                  bf16_t* __restrict__ out, int K, int N) {
    __shared__ float tile[32][33];
    const int bk = blockIdx.x * 32;
    const int bn = blockIdx.y * 32;
    const int tx = threadIdx.x & 31;
    const int ty = threadIdx.x >> 5;   // 0..7
    #pragma unroll
    for (int i = 0; i < 32; i += 8)
        tile[ty + i][tx] = in[(size_t)(bk + ty + i) * N + bn + tx];
    __syncthreads();
    #pragma unroll
    for (int i = 0; i < 32; i += 8)
        out[(size_t)(bn + ty + i) * K + bk + tx] = f2bf(tile[tx][ty + i]);
}

// ---------------------------------------------------------------------------
// bf16 WMMA GEMM, LDS-free:  C[M,N] = A[M,K] * BT[N,K]^T
//   A  : M x K row-major bf16
//   BT : N x K row-major bf16 (weights pre-transposed)
//
// Block: 256 threads = 8 waves arranged 2(M) x 4(N); wave tile 32x32
// (2x2 WMMA tiles). Per k-step: 8 global_load_b128 + 4 WMMAs.
//
// Fragment layouts per CDNA5 ISA (16-bit, wave32):
//   A (16x32):  M = lane&15; elem e -> K = (e<8?0:16) + (lane<16?0:8) + (e&7)
//   B (32x16):  N = lane&15; elem e -> K = (e<8?0:16) + (lane<16?0:8) + (e&7)
//   C/D f32:    N = lane&15; vgpr r -> M = r + (lane<16?0:8)
//
// MODE 0: QKV epilogue -> Q,K (B,H,S,D) and V transposed (B,H,D,S), bf16.
// MODE 1: proj epilogue -> fp32 row-major out.
// ---------------------------------------------------------------------------
template <int MODE>
__global__ __launch_bounds__(256)
void gemm_bf16_kernel(const bf16_t* __restrict__ A, const bf16_t* __restrict__ BT,
                      int N, int K,
                      bf16_t* __restrict__ qout, bf16_t* __restrict__ kout,
                      bf16_t* __restrict__ vout, float* __restrict__ out) {
    const int t    = threadIdx.x;
    const int lane = t & 31;
    const int wave = t >> 5;
    const int wm   = wave >> 2;                  // 0..1
    const int wn   = wave & 3;                   // 0..3
    const int m0   = blockIdx.x * 64 + wm * 32;  // wave rows m0..m0+31
    const int n0   = blockIdx.y * 128 + wn * 32; // wave cols n0..n0+31

    const unsigned short* Ag = (const unsigned short*)A;
    const unsigned short* Bg = (const unsigned short*)BT;

    const int selA = (lane < 16) ? 0 : 8;
    const int selB = (lane < 16) ? 0 : 16;

    // Per-lane source rows for the 2 A fragments and 2 B fragments
    const unsigned short* arow[2];
    const unsigned short* brow[2];
    #pragma unroll
    for (int i = 0; i < 2; ++i) {
        arow[i] = Ag + (size_t)(m0 + i * 16 + (lane & 15)) * K;
        brow[i] = Bg + (size_t)(n0 + i * 16 + (lane & 15)) * K;
    }

    v8f acc[2][2];
    #pragma unroll
    for (int a = 0; a < 2; ++a)
        #pragma unroll
        for (int b = 0; b < 2; ++b)
            #pragma unroll
            for (int i = 0; i < 8; ++i) acc[a][b][i] = 0.0f;

    for (int k0 = 0; k0 < K; k0 += 32) {
        FragBF af[2], bfm[2];
        #pragma unroll
        for (int i = 0; i < 2; ++i) {
            af[i].q[0]  = *(const uint4*)(arow[i] + k0 + selA);
            af[i].q[1]  = *(const uint4*)(arow[i] + k0 + 16 + selA);
            bfm[i].q[0] = *(const uint4*)(brow[i] + k0 + selA);
            bfm[i].q[1] = *(const uint4*)(brow[i] + k0 + 16 + selA);
        }
        // Prefetch next k-step fragments (global_prefetch_b8)
        __builtin_prefetch(arow[0] + k0 + 32, 0, 1);
        __builtin_prefetch(arow[1] + k0 + 32, 0, 1);
        __builtin_prefetch(brow[0] + k0 + 32, 0, 1);
        __builtin_prefetch(brow[1] + k0 + 32, 0, 1);
        #pragma unroll
        for (int a = 0; a < 2; ++a)
            #pragma unroll
            for (int b = 0; b < 2; ++b)
                acc[a][b] = __builtin_amdgcn_wmma_f32_16x16x32_bf16(
                    false, af[a].v, false, bfm[b].v, (short)0, acc[a][b], false, false);
    }

    // Epilogue
    const int Mbase = (lane < 16) ? 0 : 8;
    #pragma unroll
    for (int a = 0; a < 2; ++a) {
        #pragma unroll
        for (int b2 = 0; b2 < 2; ++b2) {
            const int cn = n0 + b2 * 16 + (lane & 15);
            #pragma unroll
            for (int r = 0; r < 8; ++r) {
                int row = m0 + a * 16 + Mbase + r;
                if constexpr (MODE == 0) {
                    int bb    = row >> 11;        // / SEQ
                    int s     = row & 2047;       // % SEQ
                    int which = cn >> 10;         // 0=q 1=k 2=v
                    int e     = cn & 1023;
                    int h     = e >> 6;
                    int d     = e & 63;
                    bf16_t v  = f2bf(acc[a][b2][r]);
                    int bh    = bb * HEADS + h;
                    if (which == 0)
                        qout[((size_t)bh * SEQ + s) * HDIM + d] = v;
                    else if (which == 1)
                        kout[((size_t)bh * SEQ + s) * HDIM + d] = v;
                    else // V stored transposed: (B,H,D,S)
                        vout[((size_t)bh * HDIM + d) * SEQ + s] = v;
                } else {
                    out[(size_t)row * N + cn] = acc[a][b2][r];
                }
            }
        }
    }
}

// ---------------------------------------------------------------------------
// Flash attention: one wave (32 threads) per (b,h, 16-query tile).
// Q,K bf16 (B,H,S,D); V bf16 transposed (B,H,D,S) -> every fragment is a
// pair of contiguous 16B global loads. Keys processed 32/step. fp32 online
// softmax; P relaid to A-frag form via 1KB LDS; O accumulated with 4 WMMAs.
// ---------------------------------------------------------------------------
__global__ __launch_bounds__(32)
void attn_kernel(const bf16_t* __restrict__ qb, const bf16_t* __restrict__ kb,
                 const bf16_t* __restrict__ vtb, bf16_t* __restrict__ yb) {
    __shared__ unsigned short sP[16 * 32];  // [m][key]  (P as A-frag source)

    const int lane = threadIdx.x;
    const int qt   = blockIdx.x;        // query tile  0..127
    const int bh   = blockIdx.y;        // b*HEADS + h 0..63
    const int q0   = qt * 16;
    const int h    = bh & (HEADS - 1);
    const int b    = bh >> 4;

    const unsigned short* qptr  = (const unsigned short*)qb  + ((size_t)bh * SEQ + q0) * HDIM;
    const unsigned short* kbase = (const unsigned short*)kb  + (size_t)bh * SEQ * HDIM;
    const unsigned short* vtbase= (const unsigned short*)vtb + (size_t)bh * HDIM * SEQ;

    const int m     = lane & 15;
    const int selA  = (lane < 16) ? 0 : 8;
    const int selB  = (lane < 16) ? 0 : 16;
    const int Mbase = (lane < 16) ? 0 : 8;

    // Q A-fragments (d-ktiles 0..31, 32..63): direct global 16B loads
    FragBF aq[2];
    #pragma unroll
    for (int dt = 0; dt < 2; ++dt) {
        aq[dt].q[0] = *(const uint4*)(qptr + m * HDIM + dt * 32 + selA);
        aq[dt].q[1] = *(const uint4*)(qptr + m * HDIM + dt * 32 + 16 + selA);
    }

    // Per-lane V fragment row pointers (N = d fixed per nt)
    const unsigned short* vrow[4];
    #pragma unroll
    for (int nt = 0; nt < 4; ++nt)
        vrow[nt] = vtbase + (size_t)(nt * 16 + (lane & 15)) * SEQ;

    float mrow[8], lrow[8];
    v8f accv[4];
    #pragma unroll
    for (int r = 0; r < 8; ++r) { mrow[r] = -__builtin_inff(); lrow[r] = 0.0f; }
    #pragma unroll
    for (int nt = 0; nt < 4; ++nt)
        #pragma unroll
        for (int i = 0; i < 8; ++i) accv[nt][i] = 0.0f;

    const float scale  = 0.125f;            // 1/sqrt(64)
    const int   nsteps = (q0 + 16 + 31) >> 5;

    for (int st = 0; st < nsteps; ++st) {
        const int k0 = st * 32;
        const unsigned short* kptr = kbase + (size_t)k0 * HDIM;

        // Prefetch next step's K tile rows for this lane
        __builtin_prefetch(kptr + 32 * HDIM + (lane & 15) * HDIM, 0, 1);
        __builtin_prefetch(vrow[0] + k0 + 32, 0, 1);

        // S = Q K^T : two 16-key subtiles, each 2 WMMAs over d
        v8f sacc[2];
        #pragma unroll
        for (int sub = 0; sub < 2; ++sub) {
            v8f s;
            #pragma unroll
            for (int i = 0; i < 8; ++i) s[i] = 0.0f;
            int key = sub * 16 + (lane & 15);
            #pragma unroll
            for (int dt = 0; dt < 2; ++dt) {
                FragBF bk;   // B-frag: N=key, contraction=d
                bk.q[0] = *(const uint4*)(kptr + key * HDIM + dt * 32 + selA);
                bk.q[1] = *(const uint4*)(kptr + key * HDIM + dt * 32 + 16 + selA);
                s = __builtin_amdgcn_wmma_f32_16x16x32_bf16(
                        false, aq[dt].v, false, bk.v, (short)0, s, false, false);
            }
            sacc[sub] = s;
        }

        // Online softmax (fp32). Row M = r + Mbase lives on 16 lanes (N=lane&15).
        float alpha[8];
        #pragma unroll
        for (int r = 0; r < 8; ++r) {
            int qrow = q0 + Mbase + r;
            float s0 = sacc[0][r] * scale;
            float s1 = sacc[1][r] * scale;
            int key0 = k0 + (lane & 15);
            if (key0 > qrow)      s0 = -__builtin_inff();
            if (key0 + 16 > qrow) s1 = -__builtin_inff();
            float mx = fmaxf(s0, s1);
            #pragma unroll
            for (int off = 1; off < 16; off <<= 1)
                mx = fmaxf(mx, __shfl_xor(mx, off, 16));
            float mnew = fmaxf(mrow[r], mx);
            float al   = __expf(mrow[r] - mnew);
            float p0   = __expf(s0 - mnew);
            float p1   = __expf(s1 - mnew);
            float ps   = p0 + p1;
            #pragma unroll
            for (int off = 1; off < 16; off <<= 1)
                ps += __shfl_xor(ps, off, 16);
            lrow[r]  = lrow[r] * al + ps;
            mrow[r]  = mnew;
            alpha[r] = al;
            // Store P into LDS in [m][key] form for the A-frag gather
            int M = Mbase + r;
            sP[M * 32 + (lane & 15)]      = f2us(p0);
            sP[M * 32 + 16 + (lane & 15)] = f2us(p1);
        }

        // Rescale O accumulators by alpha (row-aligned with D-layout)
        #pragma unroll
        for (int nt = 0; nt < 4; ++nt)
            #pragma unroll
            for (int r = 0; r < 8; ++r) accv[nt][r] *= alpha[r];

        __syncthreads();   // sP visible to frag gathers

        // O += P V : P as 16x32 A-frag (LDS), V as 32x16 B-frags (global)
        FragBF pf;
        pf.q[0] = *(const uint4*)&sP[m * 32 + selA];
        pf.q[1] = *(const uint4*)&sP[m * 32 + 16 + selA];
        #pragma unroll
        for (int nt = 0; nt < 4; ++nt) {
            FragBF vf;   // B-frag: N=d, contraction=key (V transposed in global)
            vf.q[0] = *(const uint4*)(vrow[nt] + k0 + selA);
            vf.q[1] = *(const uint4*)(vrow[nt] + k0 + 16 + selA);
            accv[nt] = __builtin_amdgcn_wmma_f32_16x16x32_bf16(
                           false, pf.v, false, vf.v, (short)0, accv[nt], false, false);
        }
        __syncthreads();   // protect sP before next iteration
    }

    // Normalize and store y (B,S,E) bf16: e = h*64 + nt*16 + (lane&15)
    #pragma unroll
    for (int r = 0; r < 8; ++r) {
        float inv = 1.0f / lrow[r];
        int srow  = q0 + Mbase + r;
        #pragma unroll
        for (int nt = 0; nt < 4; ++nt) {
            int e = h * HDIM + nt * 16 + (lane & 15);
            yb[((size_t)b * SEQ + srow) * EMB + e] = f2bf(accv[nt][r] * inv);
        }
    }
}

// ---------------------------------------------------------------------------
// Host-side launch
// ---------------------------------------------------------------------------
extern "C" void kernel_launch(void* const* d_in, const int* in_sizes, int n_in,
                              void* d_out, int out_size, void* d_ws, size_t ws_size,
                              hipStream_t stream) {
    const float* x      = (const float*)d_in[0];   // (B,S,E)
    const float* w_attn = (const float*)d_in[1];   // (E,3E)
    const float* w_proj = (const float*)d_in[2];   // (E,E)
    float* out = (float*)d_out;                    // (B,S,E)

    char* ws = (char*)d_ws;
    bf16_t* x_bf  = (bf16_t*)ws;  ws += (size_t)ROWS * EMB * 2;          // 16 MB
    bf16_t* waT   = (bf16_t*)ws;  ws += (size_t)EMB * (3 * EMB) * 2;     // 6 MB  [3E][E]
    bf16_t* wpT   = (bf16_t*)ws;  ws += (size_t)EMB * EMB * 2;           // 2 MB  [E][E]
    bf16_t* q_bf  = (bf16_t*)ws;  ws += (size_t)ROWS * EMB * 2;          // 16 MB (B,H,S,D)
    bf16_t* k_bf  = (bf16_t*)ws;  ws += (size_t)ROWS * EMB * 2;          // 16 MB (B,H,S,D)
    bf16_t* vT_bf = (bf16_t*)ws;  ws += (size_t)ROWS * EMB * 2;          // 16 MB (B,H,D,S)
    bf16_t* y_bf  = (bf16_t*)ws;  ws += (size_t)ROWS * EMB * 2;          // 16 MB (B,S,E)

    // 1) converts: x keeps layout; weights transpose-convert to [N][K]
    f32_to_bf16_kernel<<<dim3(4096), 256, 0, stream>>>(x, x_bf, ROWS * EMB);
    transpose_f32_to_bf16_kernel<<<dim3(EMB / 32, (3 * EMB) / 32), 256, 0, stream>>>(
        w_attn, waT, EMB, 3 * EMB);
    transpose_f32_to_bf16_kernel<<<dim3(EMB / 32, EMB / 32), 256, 0, stream>>>(
        w_proj, wpT, EMB, EMB);

    // 2) QKV GEMM: (8192 x 1024) * (1024 x 3072), scatter to Q/K/V^T
    gemm_bf16_kernel<0><<<dim3(ROWS / 64, (3 * EMB) / 128), 256, 0, stream>>>(
        x_bf, waT, 3 * EMB, EMB, q_bf, k_bf, vT_bf, nullptr);

    // 3) Flash attention: one wave per 16-query tile per (b,h)
    attn_kernel<<<dim3(SEQ / 16, BATCH * HEADS), 32, 0, stream>>>(
        q_bf, k_bf, vT_bf, y_bf);

    // 4) Output projection: (8192 x 1024) * (1024 x 1024) -> fp32 out
    gemm_bf16_kernel<1><<<dim3(ROWS / 64, EMB / 128), 256, 0, stream>>>(
        y_bf, wpT, EMB, EMB, nullptr, nullptr, nullptr, out);
}
